// QuietStarEngine_34677565948780
// MI455X (gfx1250) — matrix operand order repaired and checked by reading, not compiled, not run on hardware
//
#include <hip/hip_runtime.h>
#include <hip/hip_bf16.h>

typedef _Float16 f16_t;
typedef __attribute__((ext_vector_type(16))) _Float16 v16h;
typedef __attribute__((ext_vector_type(8)))  _Float16 v8h;
typedef __attribute__((ext_vector_type(8)))  float    v8f;

#define NEGBIG 1e30f

__device__ inline v8f wmma_f16(v16h a, v16h b, v8f c) {
  return __builtin_amdgcn_wmma_f32_16x16x32_f16(false, a, false, b, (short)0, c,
                                                false, false);
}

__device__ inline v8f vzero8() {
  v8f z;
#pragma unroll
  for (int i = 0; i < 8; ++i) z[i] = 0.0f;
  return z;
}

__device__ inline float grpmax16(float v) {
#pragma unroll
  for (int m = 1; m <= 8; m <<= 1) v = fmaxf(v, __shfl_xor(v, m, 32));
  return v;
}
__device__ inline float grpsum16(float v) {
#pragma unroll
  for (int m = 1; m <= 8; m <<= 1) v += __shfl_xor(v, m, 32);
  return v;
}

// Async 16-byte copy global -> LDS (per lane), tracked by ASYNCcnt.
__device__ inline void async_b128(unsigned lds_off, const void* gptr) {
  asm volatile("global_load_async_to_lds_b128 %0, %1, off"
               :: "v"(lds_off), "v"(gptr) : "memory");
}
__device__ inline unsigned lds_off_of(const void* p) {
  // LDS aperture flat address: low 32 bits are the in-allocation LDS offset.
  return (unsigned)(uintptr_t)p;
}

// ---------------------------------------------------------------------------
// Block-tiled GEMM:  C[M,N] = act(A[M,K] @ W[N,K]^T + bias[N])
// A, W in f16. 256 threads = 8 waves; block tile 128x128, K-step 32.
// Tiles staged in LDS with global_load_async_to_lds_b128, double buffered.
// Each wave computes 32x64 (2 M-frags x 4 N-frags, 8 WMMAs per K-step).
// M % 128 == 0 and K % 32 == 0 guaranteed by caller; N edges guarded.
// ---------------------------------------------------------------------------
#define BM 128
#define BN 128
#define BK 32
#define BKP 40   // padded row (halves) to spread LDS banks

__global__ __launch_bounds__(256) void qs_gemm_wmma(
    const f16_t* __restrict__ A, const f16_t* __restrict__ W,
    const float* __restrict__ bias, float* __restrict__ Cf,
    f16_t* __restrict__ Ch, int M, int N, int K, int relu) {
  __shared__ f16_t As[2][BM][BKP];
  __shared__ f16_t Bs[2][BN][BKP];

  const int tid  = threadIdx.x;
  const int lane = tid & 31;
  const int wid  = tid >> 5;      // 0..7
  const int l16  = lane & 15;
  const int hi   = lane >> 4;     // lane group
  const int wm   = wid >> 1;      // 0..3 (rows)
  const int wn   = wid & 1;       // 0..1 (cols)
  const int tm   = blockIdx.y;
  const int tn   = blockIdx.x;

  const int nk = K / BK;

  // Cooperative tile staging: 512 16B-chunks per tile; wave issues 2+2 ops.
  auto issue = [&](int kt, int buf) {
    const int k0 = kt * BK;
#pragma unroll
    for (int i = 0; i < 2; ++i) {
      const int chunk = wid * 64 + i * 32 + lane;   // 0..511
      const int row = chunk >> 2, sub = chunk & 3;
      async_b128(lds_off_of(&As[buf][row][sub * 8]),
                 A + (size_t)(tm * BM + row) * K + k0 + sub * 8);
    }
#pragma unroll
    for (int i = 0; i < 2; ++i) {
      const int chunk = wid * 64 + i * 32 + lane;
      const int row = chunk >> 2, sub = chunk & 3;
      int n = tn * BN + row;
      if (n > N - 1) n = N - 1;   // clamp: keep async-op count wave-uniform
      async_b128(lds_off_of(&Bs[buf][row][sub * 8]),
                 W + (size_t)n * K + k0 + sub * 8);
    }
  };

  v8f acc[2][4];
#pragma unroll
  for (int mt = 0; mt < 2; ++mt)
#pragma unroll
    for (int s = 0; s < 4; ++s) acc[mt][s] = vzero8();

  issue(0, 0);

  for (int kt = 0; kt < nk; ++kt) {
    const int buf = kt & 1;
    const bool more = (kt + 1) < nk;
    if (more) issue(kt + 1, buf ^ 1);
    if (more) asm volatile("s_wait_asynccnt 0x4" ::: "memory");
    else      asm volatile("s_wait_asynccnt 0x0" ::: "memory");
    __syncthreads();

    // A fragments (2 per wave)
    v16h af[2];
    const int akb = hi ? 8 : 0;
#pragma unroll
    for (int mt = 0; mt < 2; ++mt) {
      const f16_t* ap = &As[buf][wm * 32 + mt * 16 + l16][0];
      v8h lo = *(const v8h*)(ap + akb);
      v8h hg = *(const v8h*)(ap + akb + 16);
      v16h a;
#pragma unroll
      for (int h = 0; h < 8; ++h) { a[h] = lo[h]; a[h + 8] = hg[h]; }
      af[mt] = a;
    }
    // B fragments + 8 WMMAs
#pragma unroll
    for (int s = 0; s < 4; ++s) {
      const f16_t* bp = &Bs[buf][wn * 64 + s * 16 + l16][hi ? 8 : 0];
      v8h b0 = *(const v8h*)(bp);
      v8h b1 = *(const v8h*)(bp + 16);
      v16h b;
#pragma unroll
      for (int h = 0; h < 8; ++h) { b[h] = b0[h]; b[h + 8] = b1[h]; }
      acc[0][s] = wmma_f16(af[0], b, acc[0][s]);
      acc[1][s] = wmma_f16(af[1], b, acc[1][s]);
    }
    __syncthreads();
  }

  // Epilogue
  const int roff = hi ? 8 : 0;
#pragma unroll
  for (int s = 0; s < 4; ++s) {
    const int col = tn * BN + wn * 64 + s * 16 + l16;
    if (col >= N) continue;
    const float bv = bias ? bias[col] : 0.0f;
#pragma unroll
    for (int mt = 0; mt < 2; ++mt) {
#pragma unroll
      for (int r = 0; r < 8; ++r) {
        const int row = tm * BM + wm * 32 + mt * 16 + roff + r;
        float v = acc[mt][s][r] + bv;
        if (relu) v = fmaxf(v, 0.0f);
        if (Cf) Cf[(size_t)row * N + col] = v;
        if (Ch) Ch[(size_t)row * N + col] = (_Float16)v;
      }
    }
  }
}

// ---------------------------------------------------------------------------
// Fused MHA, flash-style online softmax, f16 in / f16 out.
// qkv per token: [q(512)|k(512)|v(512)]; head hh uses dims hh*64..+64.
// One wave per (batch, head, 16-query tile).
// ---------------------------------------------------------------------------
__global__ __launch_bounds__(32) void qs_attn_wmma(
    const f16_t* __restrict__ qkv, f16_t* __restrict__ ctx,
    int Lseq, int causal) {
  __shared__ _Float16 plds[16][32];

  const int lane = threadIdx.x;
  const int l16  = lane & 15;
  const int hi   = lane >> 4;
  const int hh   = blockIdx.y & 7;
  const int bi   = blockIdx.y >> 3;
  const int q0   = blockIdx.x * 16;
  const int NQ   = 1536;

  const f16_t* base = qkv + (size_t)bi * Lseq * NQ;

  // Q fragments (d = 0..31 and 32..63)
  v16h aq0, aq1;
  {
    const int qrow = q0 + l16;
    const int kb   = hi ? 8 : 0;
    const f16_t* qp =
        (qrow < Lseq) ? base + (size_t)qrow * NQ + hh * 64 : nullptr;
#pragma unroll
    for (int h = 0; h < 16; ++h) {
      const int kk = kb + (h < 8 ? h : h + 8);
      aq0[h] = qp ? qp[kk] : (_Float16)0.0f;
      aq1[h] = qp ? qp[32 + kk] : (_Float16)0.0f;
    }
  }

  v8f o[4];
#pragma unroll
  for (int dt = 0; dt < 4; ++dt) o[dt] = vzero8();
  float mx[8], sm[8];
#pragma unroll
  for (int r = 0; r < 8; ++r) { mx[r] = -NEGBIG; sm[r] = 0.0f; }

  const int kmax = causal ? min(Lseq, q0 + 16) : Lseq;

  for (int k0 = 0; k0 < kmax; k0 += 32) {
    v8f sv[2];
#pragma unroll
    for (int sub = 0; sub < 2; ++sub) {
      const int key = k0 + sub * 16 + l16;
      const bool kv = key < Lseq;
      const f16_t* kp = kv ? base + (size_t)key * NQ + 512 + hh * 64 : nullptr;
      const int db = hi ? 8 : 0;
      v16h bk0, bk1;
#pragma unroll
      for (int h = 0; h < 16; ++h) {
        const int dk = db + (h < 8 ? h : h + 8);
        bk0[h] = kv ? kp[dk] : (_Float16)0.0f;
        bk1[h] = kv ? kp[32 + dk] : (_Float16)0.0f;
      }
      v8f s = vzero8();
      s = wmma_f16(aq0, bk0, s);
      s = wmma_f16(aq1, bk1, s);
#pragma unroll
      for (int r = 0; r < 8; ++r) {
        const int qg = q0 + r + (hi ? 8 : 0);
        float v = s[r] * 0.125f;   // 1/sqrt(64)
        if (key >= Lseq || (causal && key > qg)) v = -NEGBIG;
        s[r] = v;
      }
      sv[sub] = s;
    }

    // online softmax over the 32-key chunk
#pragma unroll
    for (int r = 0; r < 8; ++r) {
      float tm = grpmax16(fmaxf(sv[0][r], sv[1][r]));
      float nm = fmaxf(mx[r], tm);
      float c  = __expf(mx[r] - nm);
      mx[r] = nm;
      float p0 = __expf(sv[0][r] - nm);
      float p1 = __expf(sv[1][r] - nm);
      sv[0][r] = p0; sv[1][r] = p1;
      float ts = grpsum16(p0 + p1);
      sm[r] = sm[r] * c + ts;
#pragma unroll
      for (int dt = 0; dt < 4; ++dt) o[dt][r] *= c;
    }

    // transpose P through LDS into an A-fragment
#pragma unroll
    for (int sub = 0; sub < 2; ++sub)
#pragma unroll
      for (int r = 0; r < 8; ++r)
        plds[r + (hi ? 8 : 0)][sub * 16 + l16] = (_Float16)sv[sub][r];
    __syncthreads();

    v16h pa;
    {
      const int kb = hi ? 8 : 0;
#pragma unroll
      for (int h = 0; h < 16; ++h)
        pa[h] = plds[l16][kb + (h < 8 ? h : h + 8)];
    }
    __syncthreads();

    // O += P @ V
#pragma unroll
    for (int dt = 0; dt < 4; ++dt) {
      v16h bv;
      const int d = dt * 16 + l16;
#pragma unroll
      for (int h = 0; h < 16; ++h) {
        const int key = k0 + (h < 8 ? h : h + 8) + (hi ? 8 : 0);
        bv[h] = (key < Lseq)
                    ? base[(size_t)key * NQ + 1024 + hh * 64 + d]
                    : (_Float16)0.0f;
      }
      o[dt] = wmma_f16(pa, bv, o[dt]);
    }
  }

#pragma unroll
  for (int r = 0; r < 8; ++r) {
    const int m = q0 + r + (hi ? 8 : 0);
    if (m >= Lseq) continue;
    const float inv = 1.0f / sm[r];
#pragma unroll
    for (int dt = 0; dt < 4; ++dt)
      ctx[((size_t)bi * Lseq + m) * 512 + hh * 64 + dt * 16 + l16] =
          (_Float16)(o[dt][r] * inv);
  }
}

// ---------------------------------------------------------------------------
// Residual + LayerNorm (D=512):  x = LN(x + a)*g + b, also emits f16 copy.
// ---------------------------------------------------------------------------
__global__ __launch_bounds__(256) void qs_ln(
    float* __restrict__ x, const float* __restrict__ a,
    const float* __restrict__ g, const float* __restrict__ b,
    f16_t* __restrict__ xh) {
  __shared__ float r1[256], r2[256];
  const int row = blockIdx.x, t = threadIdx.x;
  float* xr = x + (size_t)row * 512;
  const float* ar = a + (size_t)row * 512;
  float v0 = xr[t] + ar[t];
  float v1 = xr[t + 256] + ar[t + 256];
  r1[t] = v0 + v1;
  r2[t] = v0 * v0 + v1 * v1;
  __syncthreads();
  for (int s = 128; s > 0; s >>= 1) {
    if (t < s) { r1[t] += r1[t + s]; r2[t] += r2[t + s]; }
    __syncthreads();
  }
  const float mean = r1[0] * (1.0f / 512.0f);
  const float var  = r2[0] * (1.0f / 512.0f) - mean * mean;
  const float rs   = rsqrtf(var + 1e-5f);
  const float y0 = (v0 - mean) * rs * g[t] + b[t];
  const float y1 = (v1 - mean) * rs * g[t + 256] + b[t + 256];
  xr[t] = y0;
  xr[t + 256] = y1;
  xh[(size_t)row * 512 + t] = (_Float16)y0;
  xh[(size_t)row * 512 + t + 256] = (_Float16)y1;
}

// ---------------------------------------------------------------------------
// f32 -> f16 bulk conversion (weights)
// ---------------------------------------------------------------------------
__global__ __launch_bounds__(256) void qs_cvt(const float* __restrict__ s,
                                              f16_t* __restrict__ d, int n) {
  int i = blockIdx.x * 256 + threadIdx.x;
  if (i < n) d[i] = (_Float16)s[i];
}

// ---------------------------------------------------------------------------
// Thought-encoder input: x[bs,t,d] = tok_emb[ids[bs],d] + pe[s,d] + tpe[t,d]
// ---------------------------------------------------------------------------
__global__ __launch_bounds__(256) void qs_build_th(
    const float* __restrict__ tok_emb, const float* __restrict__ pe,
    const float* __restrict__ tpe, const int* __restrict__ ids,
    float* __restrict__ x, f16_t* __restrict__ xh, int total) {
  int idx = blockIdx.x * 256 + threadIdx.x;
  if (idx >= total) return;
  const int d  = idx & 511;
  const int t  = (idx >> 9) & 3;     // T = 4
  const int bs = idx >> 11;
  const int s  = bs & 255;           // S = 256
  float v = tok_emb[(size_t)ids[bs] * 512 + d] + pe[(size_t)s * 512 + d] +
            tpe[(size_t)t * 512 + d];
  x[idx] = v;
  xh[idx] = (_Float16)v;
}

// ---------------------------------------------------------------------------
// Main-encoder input splice: [base | start | th0..3 (+tpe) | end] per (b,s)
// ---------------------------------------------------------------------------
__global__ __launch_bounds__(256) void qs_build_main(
    const float* __restrict__ tok_emb, const float* __restrict__ pe,
    const float* __restrict__ tpe, const float* __restrict__ start_th,
    const float* __restrict__ end_th, const int* __restrict__ ids,
    const float* __restrict__ th, float* __restrict__ x,
    f16_t* __restrict__ xh, int total) {
  int idx = blockIdx.x * 256 + threadIdx.x;
  if (idx >= total) return;
  const int d = idx & 511;
  const int j = (idx >> 9) % 1792;   // L = S*(T+3)
  const int b = (idx >> 9) / 1792;
  const int s = j / 7;
  const int slot = j % 7;
  const int bs = b * 256 + s;
  float v;
  if (slot == 0)
    v = tok_emb[(size_t)ids[bs] * 512 + d] + pe[(size_t)s * 512 + d];
  else if (slot == 1)
    v = start_th[d];
  else if (slot == 6)
    v = end_th[d];
  else {
    const int t = slot - 2;
    v = th[((size_t)bs * 4 + t) * 512 + d] + tpe[(size_t)t * 512 + d];
  }
  x[idx] = v;
  xh[idx] = (_Float16)v;
}

// ---------------------------------------------------------------------------
// Host orchestration
// ---------------------------------------------------------------------------
static inline void run_gemm(const f16_t* A, const f16_t* W, const float* bias,
                            float* Cf, f16_t* Ch, int M, int N, int K, int relu,
                            hipStream_t st) {
  dim3 grid((N + BN - 1) / BN, M / BM);
  qs_gemm_wmma<<<grid, 256, 0, st>>>(A, W, bias, Cf, Ch, M, N, K, relu);
}
static inline void run_cvt(const float* s, f16_t* d, size_t n, hipStream_t st) {
  qs_cvt<<<(unsigned)((n + 255) / 256), 256, 0, st>>>(s, d, (int)n);
}

extern "C" void kernel_launch(void* const* d_in, const int* in_sizes, int n_in,
                              void* d_out, int out_size, void* d_ws,
                              size_t ws_size, hipStream_t stream) {
  (void)in_sizes; (void)n_in; (void)out_size; (void)ws_size;
  const float* tok_emb  = (const float*)d_in[0];
  const float* pe       = (const float*)d_in[1];
  const float* tpe      = (const float*)d_in[2];
  const float* start_th = (const float*)d_in[3];
  const float* end_th   = (const float*)d_in[4];
  const float* lm_w     = (const float*)d_in[5];
  const float* lm_b     = (const float*)d_in[6];
  const int*   ids      = (const int*)d_in[7];
  // d_in[8]: n_think_tokens (device scalar); shapes fixed at T=4.
  const float* m_qkv_w = (const float*)d_in[9];
  const float* m_qkv_b = (const float*)d_in[10];
  const float* m_out_w = (const float*)d_in[11];
  const float* m_out_b = (const float*)d_in[12];
  const float* m_w1    = (const float*)d_in[13];
  const float* m_b1    = (const float*)d_in[14];
  const float* m_w2    = (const float*)d_in[15];
  const float* m_b2    = (const float*)d_in[16];
  const float* m_ln1w  = (const float*)d_in[17];
  const float* m_ln1b  = (const float*)d_in[18];
  const float* m_ln2w  = (const float*)d_in[19];
  const float* m_ln2b  = (const float*)d_in[20];
  const float* t_qkv_w = (const float*)d_in[21];
  const float* t_qkv_b = (const float*)d_in[22];
  const float* t_out_w = (const float*)d_in[23];
  const float* t_out_b = (const float*)d_in[24];
  const float* t_w1    = (const float*)d_in[25];
  const float* t_b1    = (const float*)d_in[26];
  const float* t_w2    = (const float*)d_in[27];
  const float* t_b2    = (const float*)d_in[28];
  const float* t_ln1w  = (const float*)d_in[29];
  const float* t_ln1b  = (const float*)d_in[30];
  const float* t_ln2w  = (const float*)d_in[31];
  const float* t_ln2b  = (const float*)d_in[32];

  const int TOK_TH = 16384;  // (B*S) * T rows of thought tokens
  const int TOK_MN = 7168;   // B * 1792

  char* p = (char*)d_ws;
  auto alloc_f = [&](size_t n) { float* r = (float*)p; p += n * 4; return r; };
  auto alloc_h = [&](size_t n) { f16_t* r = (f16_t*)p; p += n * 2; return r; };

  float* x_th_f = alloc_f((size_t)TOK_TH * 512);
  float* x_mn_f = alloc_f((size_t)TOK_MN * 512);
  float* tmpb   = alloc_f((size_t)TOK_TH * 512);
  f16_t* x_th_h = alloc_h((size_t)TOK_TH * 512);
  f16_t* x_mn_h = alloc_h((size_t)TOK_MN * 512);
  f16_t* qkv_h  = alloc_h((size_t)TOK_TH * 1536);
  f16_t* ctx_h  = alloc_h((size_t)TOK_TH * 512);
  f16_t* ffn_h  = alloc_h((size_t)TOK_TH * 2048);
  f16_t* wm_qkv = alloc_h((size_t)6 * 1536 * 512);
  f16_t* wm_out = alloc_h((size_t)6 * 512 * 512);
  f16_t* wm_w1  = alloc_h((size_t)6 * 2048 * 512);
  f16_t* wm_w2  = alloc_h((size_t)6 * 512 * 2048);
  f16_t* wt_qkv = alloc_h((size_t)4 * 1536 * 512);
  f16_t* wt_out = alloc_h((size_t)4 * 512 * 512);
  f16_t* wt_w1  = alloc_h((size_t)4 * 2048 * 512);
  f16_t* wt_w2  = alloc_h((size_t)4 * 512 * 2048);
  f16_t* wlm    = alloc_h((size_t)10000 * 512);

  // ---- weights to f16 (once per launch) ----
  run_cvt(m_qkv_w, wm_qkv, (size_t)6 * 1536 * 512, stream);
  run_cvt(m_out_w, wm_out, (size_t)6 * 512 * 512, stream);
  run_cvt(m_w1,    wm_w1,  (size_t)6 * 2048 * 512, stream);
  run_cvt(m_w2,    wm_w2,  (size_t)6 * 512 * 2048, stream);
  run_cvt(t_qkv_w, wt_qkv, (size_t)4 * 1536 * 512, stream);
  run_cvt(t_out_w, wt_out, (size_t)4 * 512 * 512, stream);
  run_cvt(t_w1,    wt_w1,  (size_t)4 * 2048 * 512, stream);
  run_cvt(t_w2,    wt_w2,  (size_t)4 * 512 * 2048, stream);
  run_cvt(lm_w,    wlm,    (size_t)10000 * 512, stream);

  // ---- thought encoder input ----
  {
    int total = TOK_TH * 512;
    qs_build_th<<<(total + 255) / 256, 256, 0, stream>>>(
        tok_emb, pe, tpe, ids, x_th_f, x_th_h, total);
  }
  // ---- thought encoder: 4 layers, seq=4, 4096 batches, no mask ----
  for (int l = 0; l < 4; ++l) {
    run_gemm(x_th_h, wt_qkv + (size_t)l * 1536 * 512, t_qkv_b + (size_t)l * 1536,
             nullptr, qkv_h, TOK_TH, 1536, 512, 0, stream);
    qs_attn_wmma<<<dim3(1, 4096 * 8), 32, 0, stream>>>(qkv_h, ctx_h, 4, 0);
    run_gemm(ctx_h, wt_out + (size_t)l * 512 * 512, t_out_b + (size_t)l * 512,
             tmpb, nullptr, TOK_TH, 512, 512, 0, stream);
    qs_ln<<<TOK_TH, 256, 0, stream>>>(x_th_f, tmpb, t_ln1w + (size_t)l * 512,
                                      t_ln1b + (size_t)l * 512, x_th_h);
    run_gemm(x_th_h, wt_w1 + (size_t)l * 2048 * 512, t_b1 + (size_t)l * 2048,
             nullptr, ffn_h, TOK_TH, 2048, 512, 1, stream);
    run_gemm(ffn_h, wt_w2 + (size_t)l * 512 * 2048, t_b2 + (size_t)l * 512,
             tmpb, nullptr, TOK_TH, 512, 2048, 0, stream);
    qs_ln<<<TOK_TH, 256, 0, stream>>>(x_th_f, tmpb, t_ln2w + (size_t)l * 512,
                                      t_ln2b + (size_t)l * 512, x_th_h);
  }

  // ---- splice into main sequence ----
  {
    int total = TOK_MN * 512;
    qs_build_main<<<(total + 255) / 256, 256, 0, stream>>>(
        tok_emb, pe, tpe, start_th, end_th, ids, x_th_f, x_mn_f, x_mn_h, total);
  }
  // ---- main encoder: 6 layers, seq=1792, 4 batches, causal ----
  for (int l = 0; l < 6; ++l) {
    run_gemm(x_mn_h, wm_qkv + (size_t)l * 1536 * 512, m_qkv_b + (size_t)l * 1536,
             nullptr, qkv_h, TOK_MN, 1536, 512, 0, stream);
    qs_attn_wmma<<<dim3(112, 4 * 8), 32, 0, stream>>>(qkv_h, ctx_h, 1792, 1);
    run_gemm(ctx_h, wm_out + (size_t)l * 512 * 512, m_out_b + (size_t)l * 512,
             tmpb, nullptr, TOK_MN, 512, 512, 0, stream);
    qs_ln<<<TOK_MN, 256, 0, stream>>>(x_mn_f, tmpb, m_ln1w + (size_t)l * 512,
                                      m_ln1b + (size_t)l * 512, x_mn_h);
    run_gemm(x_mn_h, wm_w1 + (size_t)l * 2048 * 512, m_b1 + (size_t)l * 2048,
             nullptr, ffn_h, TOK_MN, 2048, 512, 1, stream);
    run_gemm(ffn_h, wm_w2 + (size_t)l * 512 * 2048, m_b2 + (size_t)l * 512,
             tmpb, nullptr, TOK_MN, 512, 2048, 0, stream);
    qs_ln<<<TOK_MN, 256, 0, stream>>>(x_mn_f, tmpb, m_ln2w + (size_t)l * 512,
                                      m_ln2b + (size_t)l * 512, x_mn_h);
  }

  // ---- LM head: [7168,512]f16 @ [10000,512]f16^T + lm_b -> d_out f32 ----
  run_gemm(x_mn_h, wlm, lm_b, (float*)d_out, nullptr, TOK_MN, 10000, 512, 0,
           stream);
}
